// CSA_66030827208858
// MI455X (gfx1250) — compile-verified, matrix operand order checked by reading, not run on hardware
//
#include <hip/hip_runtime.h>
#include <hip/hip_bf16.h>
#include <math.h>
#include <stdint.h>

typedef __attribute__((ext_vector_type(16))) __bf16 v16bf;
typedef __attribute__((ext_vector_type(8)))  float  v8f;

#define T_    8192
#define HID_  2048
#define H_    16
#define HD_   128
#define QC_   512
#define NB_   512
#define MB_   16
#define CI_   64
#define NIH_  4
#define TOPK_ 64
#define WIN_  256
#define G_    4
#define DG_   512
#define NEGV  (-1.0e30f)
#define EPS_  1e-6f
#define SCALE_ 0.08838834764831845f  /* 1/sqrt(128) */

// ---------- helpers ----------
__device__ __forceinline__ int kmap(int v, int half) {
  // 16-bit WMMA A/B K index for VGPR v, lane-half (ISA 7.12.2)
  return (v < 4 ? 2 * v : 16 + 2 * (v - 4)) + 8 * half;
}

// CDNA5 async global->LDS copy (VGLOBAL op 98), tracked by ASYNCcnt.
__device__ __forceinline__ void async_ld_b128(unsigned lds_addr, const void* gptr) {
  asm volatile("global_load_async_to_lds_b128 %0, %1, off"
               : : "v"(lds_addr), "v"(gptr) : "memory");
}
__device__ __forceinline__ void wait_async0() {
  asm volatile("s_wait_asynccnt 0" : : : "memory");
}
__device__ __forceinline__ unsigned lds_off32(const void* p) {
  // generic pointer to LDS: low 32 bits are the LDS byte offset
  return (unsigned)(uintptr_t)p;
}

__global__ void k_pack_bf16(const float* __restrict__ src, __bf16* __restrict__ dst, int n) {
  int i = blockIdx.x * blockDim.x + threadIdx.x;
  if (i < n) dst[i] = (__bf16)src[i];
}

// ---------- generic WMMA GEMM: C[M,N] = A[M,K] * B[K,N], bf16 in, f32 out ----------
__global__ __launch_bounds__(256) void k_gemm_bf16(
    const __bf16* __restrict__ A, const __bf16* __restrict__ B, float* __restrict__ C,
    int M, int N, int K, int lda, int ldb, int ldc)
{
  __shared__ __align__(16) __bf16 sA[64][32];
  __shared__ __align__(16) __bf16 sBt[64][36];  // B transposed tile: [n][k], padded
  int tid = threadIdx.x;
  int wave = tid >> 5, lane = tid & 31;
  int half = lane >> 4, l15 = lane & 15;
  int bm = blockIdx.y * 64, bn = blockIdx.x * 64;
  int cg = wave & 3, rg = wave >> 2;
  int m0 = rg * 32, m1 = m0 + 16, n0 = cg * 16;
  // A staging: one 16B chunk per thread per k-step, via async-to-LDS
  int ar = tid >> 2, ac = (tid & 3) * 8;
  unsigned aLds = lds_off32(&sA[ar][ac]);
  v8f acc0 = {}, acc1 = {};
  for (int k0 = 0; k0 < K; k0 += 32) {
    __syncthreads();
    async_ld_b128(aLds, A + (long)(bm + ar) * lda + k0 + ac);
#pragma unroll
    for (int e = 0; e < 8; ++e) {           // B tile: 32x64 -> transposed store
      int idx = tid * 8 + e;
      int r = idx >> 6, c = idx & 63;
      sBt[c][r] = B[(long)(k0 + r) * ldb + bn + c];
    }
    if (k0 + 32 < K) __builtin_prefetch(&A[(long)(bm + (tid >> 3)) * lda + k0 + 32], 0, 0);
    wait_async0();
    __syncthreads();
    v16bf a0, a1, bf;
#pragma unroll
    for (int v = 0; v < 8; ++v) {
      int k = kmap(v, half);
      a0[2 * v] = sA[m0 + l15][k]; a0[2 * v + 1] = sA[m0 + l15][k + 1];
      a1[2 * v] = sA[m1 + l15][k]; a1[2 * v + 1] = sA[m1 + l15][k + 1];
      bf[2 * v] = sBt[n0 + l15][k]; bf[2 * v + 1] = sBt[n0 + l15][k + 1];
    }
    acc0 = __builtin_amdgcn_wmma_f32_16x16x32_bf16(false, a0, false, bf, (short)0, acc0, false, false);
    acc1 = __builtin_amdgcn_wmma_f32_16x16x32_bf16(false, a1, false, bf, (short)0, acc1, false, false);
  }
#pragma unroll
  for (int v = 0; v < 8; ++v) {
    int r0 = bm + m0 + v + 8 * half;
    int r1 = bm + m1 + v + 8 * half;
    int cc = bn + n0 + l15;
    C[(long)r0 * ldc + cc] = acc0[v];
    C[(long)r1 * ldc + cc] = acc1[v];
  }
}

// ---------- compress: out[n][d] = sum_m softmax_m(z[n*16+m][d]+bias[m][d]) * c[n*16+m][d] ----------
__global__ void k_compress(const float* __restrict__ c, const float* __restrict__ z,
                           const float* __restrict__ bias, float* __restrict__ out, int D) {
  int i = blockIdx.x * blockDim.x + threadIdx.x;
  if (i >= NB_ * D) return;
  int n = i / D, d = i % D;
  float zb[MB_]; float mx = -INFINITY;
#pragma unroll
  for (int m = 0; m < MB_; ++m) { zb[m] = z[(long)(n * MB_ + m) * D + d] + bias[m * D + d]; mx = fmaxf(mx, zb[m]); }
  float s = 0.f;
#pragma unroll
  for (int m = 0; m < MB_; ++m) { zb[m] = __expf(zb[m] - mx); s += zb[m]; }
  float acc = 0.f;
#pragma unroll
  for (int m = 0; m < MB_; ++m) acc += (zb[m] / s) * c[(long)(n * MB_ + m) * D + d];
  out[(long)n * D + d] = acc;
}

// ---------- q: rope then rmsnorm; output transposed to [h][t][d] (f32 + bf16) ----------
__global__ __launch_bounds__(128) void k_q_rope_norm(const float* __restrict__ qraw, const float* __restrict__ qw,
                                                     float* __restrict__ qn, __bf16* __restrict__ qbf) {
  int t = blockIdx.x, h = blockIdx.y, d = threadIdx.x;
  __shared__ float sx[HD_]; __shared__ float red[4];
  float x = qraw[((long)t * H_ + h) * HD_ + d];
  sx[d] = x; __syncthreads();
  float y;
  if (d < 32) {
    float inv = __powf(10000.f, -((float)(2 * d) / 64.f));
    float ang = (float)t * inv;
    y = sx[d] * __cosf(ang) - sx[d + 32] * __sinf(ang);
  } else if (d < 64) {
    int i = d - 32;
    float inv = __powf(10000.f, -((float)(2 * i) / 64.f));
    float ang = (float)t * inv;
    y = sx[i] * __sinf(ang) + sx[d] * __cosf(ang);
  } else y = x;
  float sq = y * y;
  for (int off = 16; off; off >>= 1) sq += __shfl_xor(sq, off, 32);
  if ((d & 31) == 0) red[d >> 5] = sq;
  __syncthreads();
  float tot = red[0] + red[1] + red[2] + red[3];
  float o = y * rsqrtf(tot / 128.f + EPS_) * qw[d];
  long off = ((long)h * T_ + t) * HD_ + d;
  qn[off] = o; qbf[off] = (__bf16)o;
}

// ---------- k: rmsnorm then rope ----------
__global__ __launch_bounds__(128) void k_k_norm_rope(const float* __restrict__ kraw, const float* __restrict__ kw,
                                                     __bf16* __restrict__ swkbf) {
  int t = blockIdx.x, d = threadIdx.x;
  __shared__ float sx[HD_]; __shared__ float red[4];
  float x = kraw[(long)t * HD_ + d];
  float sq = x * x;
  for (int off = 16; off; off >>= 1) sq += __shfl_xor(sq, off, 32);
  if ((d & 31) == 0) red[d >> 5] = sq;
  __syncthreads();
  float tot = red[0] + red[1] + red[2] + red[3];
  float nx = x * rsqrtf(tot / 128.f + EPS_) * kw[d];
  sx[d] = nx; __syncthreads();
  float y;
  if (d < 32) {
    float inv = __powf(10000.f, -((float)(2 * d) / 64.f));
    float ang = (float)t * inv;
    y = sx[d] * __cosf(ang) - sx[d + 32] * __sinf(ang);
  } else if (d < 64) {
    int i = d - 32;
    float inv = __powf(10000.f, -((float)(2 * i) / 64.f));
    float ang = (float)t * inv;
    y = sx[i] * __sinf(ang) + sx[d] * __cosf(ang);
  } else y = nx;
  swkbf[(long)t * HD_ + d] = (__bf16)y;
}

// ---------- rmsnorm rows (compressed kv) ----------
__global__ __launch_bounds__(128) void k_rmsnorm_rows(const float* __restrict__ in, const float* __restrict__ w,
                                                      float* __restrict__ out) {
  int r = blockIdx.x, d = threadIdx.x;
  __shared__ float red[4];
  float x = in[(long)r * HD_ + d];
  float sq = x * x;
  for (int off = 16; off; off >>= 1) sq += __shfl_xor(sq, off, 32);
  if ((d & 31) == 0) red[d >> 5] = sq;
  __syncthreads();
  float tot = red[0] + red[1] + red[2] + red[3];
  out[(long)r * HD_ + d] = x * rsqrtf(tot / HD_ + EPS_) * w[d];
}

// ---------- wih = h @ w_w (N=4, too narrow for WMMA) ----------
__global__ void k_wih(const float* __restrict__ h, const float* __restrict__ ww, float* __restrict__ wih) {
  int i = blockIdx.x * blockDim.x + threadIdx.x;
  if (i >= T_ * NIH_) return;
  int t = i >> 2, j = i & 3;
  float acc = 0.f;
  for (int k = 0; k < HID_; ++k) acc += h[(long)t * HID_ + k] * ww[k * NIH_ + j];
  wih[i] = acc;
}

// ---------- indexer scores ----------
__global__ void k_iscores(const float* __restrict__ qi, const float* __restrict__ kcomp,
                          const float* __restrict__ wih, float* __restrict__ isc) {
  int i = blockIdx.x * blockDim.x + threadIdx.x;
  if (i >= T_ * NB_) return;
  int t = i / NB_, n = i % NB_;
  if (n * MB_ + (MB_ - 1) >= t) { isc[i] = -INFINITY; return; }
  float s = 0.f;
  for (int ih = 0; ih < NIH_; ++ih) {
    const float* qp = qi + ((long)t * NIH_ + ih) * CI_;
    const float* kp = kcomp + (long)n * CI_;
    float dot = 0.f;
    for (int c = 0; c < CI_; ++c) dot += qp[c] * kp[c];
    s += wih[t * NIH_ + ih] * fmaxf(dot, 0.f);
  }
  isc[i] = s;
}

// ---------- top-k (iterative argmax over 512) ----------
__global__ __launch_bounds__(256) void k_topk(const float* __restrict__ isc, int* __restrict__ topidx,
                                              int* __restrict__ topvalid) {
  int t = blockIdx.x, tid = threadIdx.x;
  __shared__ float sv[NB_]; __shared__ float rv[256]; __shared__ int ri[256];
  sv[tid] = isc[(long)t * NB_ + tid];
  sv[tid + 256] = isc[(long)t * NB_ + tid + 256];
  __syncthreads();
  for (int k = 0; k < TOPK_; ++k) {
    float v0 = sv[tid], v1 = sv[tid + 256];
    if (v1 > v0) { rv[tid] = v1; ri[tid] = tid + 256; } else { rv[tid] = v0; ri[tid] = tid; }
    __syncthreads();
    for (int s = 128; s; s >>= 1) {
      if (tid < s && rv[tid + s] > rv[tid]) { rv[tid] = rv[tid + s]; ri[tid] = ri[tid + s]; }
      __syncthreads();
    }
    if (tid == 0) {
      topidx[(long)t * TOPK_ + k] = ri[0];
      topvalid[(long)t * TOPK_ + k] = (rv[0] > -3.0e38f) ? 1 : 0;
      sv[ri[0]] = -INFINITY;
    }
    __syncthreads();
  }
}

// ---------- sparse attention over top-64 compressed kv ----------
__global__ __launch_bounds__(128) void k_sparse_attn(const float* __restrict__ qn, const float* __restrict__ kvn,
                                                     const int* __restrict__ topidx, const int* __restrict__ topvalid,
                                                     float* __restrict__ attn) {
  int t = blockIdx.x, h = blockIdx.y, d = threadIdx.x;
  __shared__ float skv[TOPK_][HD_]; __shared__ float sq[HD_]; __shared__ float sc[TOPK_];
  __shared__ float sden; __shared__ int sany;
  sq[d] = qn[((long)h * T_ + t) * HD_ + d];
  for (int k = 0; k < TOPK_; ++k) {
    int idx = topidx[(long)t * TOPK_ + k]; if (idx < 0) idx = 0;
    skv[k][d] = kvn[(long)idx * HD_ + d];
  }
  __syncthreads();
  if (d < TOPK_) {
    float dot = 0.f;
    for (int c = 0; c < HD_; ++c) dot += sq[c] * skv[d][c];
    sc[d] = topvalid[(long)t * TOPK_ + d] ? dot * SCALE_ : NEGV;
  }
  __syncthreads();
  if (d == 0) {
    float mx = -INFINITY;
    for (int k = 0; k < TOPK_; ++k) mx = fmaxf(mx, sc[k]);
    float s = 0.f;
    for (int k = 0; k < TOPK_; ++k) { sc[k] = __expf(sc[k] - mx); s += sc[k]; }
    sden = s;
    int any = 0;
    for (int k = 0; k < TOPK_; ++k) any |= topvalid[(long)t * TOPK_ + k];
    sany = any;
  }
  __syncthreads();
  float inv = (sden > 0.f) ? 1.f / sden : 0.f;
  float o = 0.f;
  for (int k = 0; k < TOPK_; ++k) o += sc[k] * inv * skv[k][d];
  attn[((long)h * T_ + t) * HD_ + d] = sany ? o : 0.f;
}

// ---------- sliding-window attention (flash, WMMA, async K/V staging) ----------
__global__ __launch_bounds__(256) void k_sw_attn(const __bf16* __restrict__ qbf, const __bf16* __restrict__ kbf,
                                                 const __bf16* __restrict__ vbf, float* __restrict__ attn) {
  __shared__ __align__(16) __bf16 sK[64][HD_ + 8];
  __shared__ __align__(16) __bf16 sV[64][HD_ + 8];
  __shared__ __align__(16) __bf16 sP[8][16][64 + 8];
  int h = blockIdx.y;
  int qstart = blockIdx.x * 128;
  int nwin = qstart / WIN_;
  int tid = threadIdx.x, wave = tid >> 5, lane = tid & 31, half = lane >> 4, l15 = lane & 15;
  int qrow_base = qstart + wave * 16;
  // preload Q A-fragments (K = 128 -> 4 k-steps)
  v16bf aq[4];
  {
    long qb = ((long)h * T_ + qrow_base + l15) * HD_;
#pragma unroll
    for (int s = 0; s < 4; ++s)
#pragma unroll
      for (int v = 0; v < 8; ++v) {
        int k = kmap(v, half);
        aq[s][2 * v] = qbf[qb + s * 32 + k];
        aq[s][2 * v + 1] = qbf[qb + s * 32 + k + 1];
      }
  }
  v8f zero = {};
  v8f acc[8];
#pragma unroll
  for (int i = 0; i < 8; ++i) acc[i] = zero;
  float m[8], l[8];
#pragma unroll
  for (int v = 0; v < 8; ++v) { m[v] = -INFINITY; l[v] = 0.f; }

  for (int c = 0; c < 8; ++c) {            // 8 chunks of 64 keys = 512-key window
    __syncthreads();
    int kg0 = nwin * WIN_ - WIN_ + c * 64;
    // stage 64x128 bf16 K and V chunks via async global->LDS (16B granules)
#pragma unroll
    for (int it = 0; it < 4; ++it) {
      int e = tid + it * 256;              // e in [0, 1024): 64 rows x 16 granules
      int j = e >> 4, d8 = (e & 15) * 8;
      int kg = kg0 + j;
      if (kg >= 0) {
        async_ld_b128(lds_off32(&sK[j][d8]), kbf + (long)kg * HD_ + d8);
        async_ld_b128(lds_off32(&sV[j][d8]), vbf + (long)kg * HD_ + d8);
      } else {
#pragma unroll
        for (int u = 0; u < 8; ++u) { sK[j][d8 + u] = (__bf16)0.f; sV[j][d8 + u] = (__bf16)0.f; }
      }
    }
    wait_async0();
    __syncthreads();
    // S = Q * K^T for 4 key-tiles
    v8f S[4];
#pragma unroll
    for (int nt = 0; nt < 4; ++nt) {
      v8f sacc = zero;
#pragma unroll
      for (int s = 0; s < 4; ++s) {
        v16bf bf;
#pragma unroll
        for (int v = 0; v < 8; ++v) {
          int k = kmap(v, half);
          bf[2 * v] = sK[nt * 16 + l15][s * 32 + k];
          bf[2 * v + 1] = sK[nt * 16 + l15][s * 32 + k + 1];
        }
        sacc = __builtin_amdgcn_wmma_f32_16x16x32_bf16(false, aq[s], false, bf, (short)0, sacc, false, false);
      }
      S[nt] = sacc;
    }
    // mask + online softmax (row stats via half-wave shfl reductions)
    float mnew[8];
#pragma unroll
    for (int v = 0; v < 8; ++v) {
      int r = v + 8 * half;
      int ql = (qrow_base - nwin * WIN_) + r;
      float rowm = m[v];
#pragma unroll
      for (int nt = 0; nt < 4; ++nt) {
        int jidx = c * 64 + nt * 16 + l15;
        int diff = jidx - ql;
        bool ok = (diff > 0) && (diff <= WIN_) && (nwin > 0 || jidx >= WIN_);
        float val = ok ? S[nt][v] * SCALE_ : NEGV;
        S[nt][v] = val;
        rowm = fmaxf(rowm, val);
      }
#pragma unroll
      for (int off = 1; off < 16; off <<= 1) rowm = fmaxf(rowm, __shfl_xor(rowm, off, 32));
      mnew[v] = rowm;
    }
#pragma unroll
    for (int v = 0; v < 8; ++v) {
      float scl = __expf(m[v] - mnew[v]);
      float rs = 0.f;
#pragma unroll
      for (int nt = 0; nt < 4; ++nt) { float p = __expf(S[nt][v] - mnew[v]); S[nt][v] = p; rs += p; }
#pragma unroll
      for (int off = 1; off < 16; off <<= 1) rs += __shfl_xor(rs, off, 32);
      l[v] = l[v] * scl + rs;
      m[v] = mnew[v];
#pragma unroll
      for (int nt = 0; nt < 8; ++nt) acc[nt][v] *= scl;
    }
    // C-layout P -> LDS -> A-fragments (per-wave buffer, wave-internal ordering via DScnt)
#pragma unroll
    for (int nt = 0; nt < 4; ++nt)
#pragma unroll
      for (int v = 0; v < 8; ++v)
        sP[wave][v + 8 * half][nt * 16 + l15] = (__bf16)S[nt][v];
#pragma unroll
    for (int s = 0; s < 2; ++s) {
      v16bf ap;
#pragma unroll
      for (int v = 0; v < 8; ++v) {
        int k = kmap(v, half);
        ap[2 * v] = sP[wave][l15][s * 32 + k];
        ap[2 * v + 1] = sP[wave][l15][s * 32 + k + 1];
      }
#pragma unroll
      for (int nt = 0; nt < 8; ++nt) {
        v16bf bv;
#pragma unroll
        for (int v = 0; v < 8; ++v) {
          int k = kmap(v, half);
          bv[2 * v] = sV[s * 32 + k][nt * 16 + l15];
          bv[2 * v + 1] = sV[s * 32 + k + 1][nt * 16 + l15];
        }
        acc[nt] = __builtin_amdgcn_wmma_f32_16x16x32_bf16(false, ap, false, bv, (short)0, acc[nt], false, false);
      }
    }
  }
  // epilogue: attn += O / l
#pragma unroll
  for (int v = 0; v < 8; ++v) {
    float invl = (l[v] > 0.f) ? 1.f / l[v] : 0.f;
    int tq = qrow_base + v + 8 * half;
#pragma unroll
    for (int nt = 0; nt < 8; ++nt) {
      long off = ((long)h * T_ + tq) * HD_ + nt * 16 + l15;
      attn[off] += acc[nt][v] * invl;
    }
  }
}

// ---------- transpose attn [h][t][d] -> o_bf [t][h*128+d] ----------
__global__ void k_attn_to_obf(const float* __restrict__ attn, __bf16* __restrict__ obf) {
  long i = (long)blockIdx.x * blockDim.x + threadIdx.x;
  if (i >= (long)T_ * HID_) return;
  int t = (int)(i / HID_); int col = (int)(i % HID_);
  int h = col >> 7, d = col & 127;
  obf[i] = (__bf16)attn[((long)h * T_ + t) * HD_ + d];
}

// =======================================================================
extern "C" void kernel_launch(void* const* d_in, const int* in_sizes, int n_in,
                              void* d_out, int out_size, void* d_ws, size_t ws_size,
                              hipStream_t stream) {
  const float* h       = (const float*)d_in[0];
  const float* w_qc    = (const float*)d_in[1];
  const float* w_qup   = (const float*)d_in[2];
  const float* kvc_w   = (const float*)d_in[3];
  const float* kvc_wz  = (const float*)d_in[4];
  const float* kvc_b   = (const float*)d_in[5];
  const float* k_proj  = (const float*)d_in[6];
  const float* v_proj  = (const float*)d_in[7];
  const float* idx_w   = (const float*)d_in[8];
  const float* idx_wz  = (const float*)d_in[9];
  const float* idx_b   = (const float*)d_in[10];
  const float* w_dq    = (const float*)d_in[11];
  const float* w_iuq   = (const float*)d_in[12];
  const float* w_w     = (const float*)d_in[13];
  const float* q_nw    = (const float*)d_in[14];
  const float* k_nw    = (const float*)d_in[15];
  const float* group_w = (const float*)d_in[16];
  const float* final_w = (const float*)d_in[17];
  float* out = (float*)d_out;

  char* base = (char*)d_ws; size_t off = 0;
  auto alloc = [&](size_t bytes) -> void* {
    off = (off + 255) & ~(size_t)255;
    void* p = base + off; off += bytes; return p;
  };
  const long nH = (long)T_ * HID_;

  __bf16* h_bf    = (__bf16*)alloc(nH * 2);
  __bf16* wqc_bf  = (__bf16*)alloc((long)HID_ * QC_ * 2);
  __bf16* wqup_bf = (__bf16*)alloc((long)QC_ * HID_ * 2);
  __bf16* kvcw_bf = (__bf16*)alloc((long)HID_ * HD_ * 2);
  __bf16* kvcz_bf = (__bf16*)alloc((long)HID_ * HD_ * 2);
  __bf16* kprj_bf = (__bf16*)alloc((long)HID_ * HD_ * 2);
  __bf16* vprj_bf = (__bf16*)alloc((long)HID_ * HD_ * 2);
  __bf16* idxw_bf = (__bf16*)alloc((long)HID_ * CI_ * 2);
  __bf16* idxz_bf = (__bf16*)alloc((long)HID_ * CI_ * 2);
  __bf16* wdq_bf  = (__bf16*)alloc((long)HID_ * CI_ * 2);
  __bf16* wiuq_bf = (__bf16*)alloc((long)CI_ * NIH_ * CI_ * 2);
  __bf16* gw_bf   = (__bf16*)alloc((long)G_ * DG_ * DG_ * 2);
  __bf16* fw_bf   = (__bf16*)alloc((long)HID_ * HID_ * 2);

  float*  qc      = (float*)alloc((long)T_ * QC_ * 4);
  __bf16* qc_bf   = (__bf16*)alloc((long)T_ * QC_ * 2);
  float*  qraw    = (float*)alloc(nH * 4);
  float*  qn      = (float*)alloc(nH * 4);
  __bf16* q_bf    = (__bf16*)alloc(nH * 2);
  float*  kvcc    = (float*)alloc((long)T_ * HD_ * 4);
  float*  kvcz    = (float*)alloc((long)T_ * HD_ * 4);
  float*  kraw    = (float*)alloc((long)T_ * HD_ * 4);
  __bf16* swk_bf  = (__bf16*)alloc((long)T_ * HD_ * 2);
  float*  swv     = (float*)alloc((long)T_ * HD_ * 4);
  __bf16* swv_bf  = (__bf16*)alloc((long)T_ * HD_ * 2);
  float*  idxc    = (float*)alloc((long)T_ * CI_ * 4);
  float*  idxz    = (float*)alloc((long)T_ * CI_ * 4);
  float*  kcomp   = (float*)alloc((long)NB_ * CI_ * 4);
  float*  ckv     = (float*)alloc((long)NB_ * HD_ * 4);
  float*  kvn     = (float*)alloc((long)NB_ * HD_ * 4);
  float*  dq      = (float*)alloc((long)T_ * CI_ * 4);
  __bf16* dq_bf   = (__bf16*)alloc((long)T_ * CI_ * 2);
  float*  qi      = (float*)alloc((long)T_ * NIH_ * CI_ * 4);
  float*  wih     = (float*)alloc((long)T_ * NIH_ * 4);
  float*  isc     = (float*)alloc((long)T_ * NB_ * 4);
  int*    topidx  = (int*)alloc((long)T_ * TOPK_ * 4);
  int*    topval  = (int*)alloc((long)T_ * TOPK_ * 4);
  float*  attn    = (float*)alloc(nH * 4);
  __bf16* o_bf    = (__bf16*)alloc(nH * 2);
  float*  og      = (float*)alloc(nH * 4);
  __bf16* og_bf   = (__bf16*)alloc(nH * 2);
  (void)ws_size; (void)in_sizes; (void)n_in; (void)out_size;

#define PACK(src, dst, n) k_pack_bf16<<<((int)(n) + 255) / 256, 256, 0, stream>>>((src), (dst), (int)(n))
#define GEMM(A, B, C, M, N, K, lda, ldb, ldc) \
  k_gemm_bf16<<<dim3((N) / 64, (M) / 64), 256, 0, stream>>>((A), (B), (C), (M), (N), (K), (lda), (ldb), (ldc))

  // bf16 packs
  PACK(h, h_bf, nH);
  PACK(w_qc, wqc_bf, HID_ * QC_);
  PACK(w_qup, wqup_bf, QC_ * HID_);
  PACK(kvc_w, kvcw_bf, HID_ * HD_);
  PACK(kvc_wz, kvcz_bf, HID_ * HD_);
  PACK(k_proj, kprj_bf, HID_ * HD_);
  PACK(v_proj, vprj_bf, HID_ * HD_);
  PACK(idx_w, idxw_bf, HID_ * CI_);
  PACK(idx_wz, idxz_bf, HID_ * CI_);
  PACK(w_dq, wdq_bf, HID_ * CI_);
  PACK(w_iuq, wiuq_bf, CI_ * NIH_ * CI_);
  PACK(group_w, gw_bf, G_ * DG_ * DG_);
  PACK(final_w, fw_bf, HID_ * HID_);

  // projections
  GEMM(h_bf, wqc_bf, qc, T_, QC_, HID_, HID_, QC_, QC_);
  PACK(qc, qc_bf, (long)T_ * QC_);
  GEMM(qc_bf, wqup_bf, qraw, T_, HID_, QC_, QC_, HID_, HID_);
  GEMM(h_bf, kvcw_bf, kvcc, T_, HD_, HID_, HID_, HD_, HD_);
  GEMM(h_bf, kvcz_bf, kvcz, T_, HD_, HID_, HID_, HD_, HD_);
  GEMM(h_bf, kprj_bf, kraw, T_, HD_, HID_, HID_, HD_, HD_);
  GEMM(h_bf, vprj_bf, swv, T_, HD_, HID_, HID_, HD_, HD_);
  GEMM(h_bf, idxw_bf, idxc, T_, CI_, HID_, HID_, CI_, CI_);
  GEMM(h_bf, idxz_bf, idxz, T_, CI_, HID_, HID_, CI_, CI_);
  GEMM(h_bf, wdq_bf, dq, T_, CI_, HID_, HID_, CI_, CI_);
  PACK(dq, dq_bf, (long)T_ * CI_);
  GEMM(dq_bf, wiuq_bf, qi, T_, NIH_ * CI_, CI_, CI_, NIH_ * CI_, NIH_ * CI_);
  k_wih<<<(T_ * NIH_ + 255) / 256, 256, 0, stream>>>(h, w_w, wih);

  // rope / norms
  k_q_rope_norm<<<dim3(T_, H_), 128, 0, stream>>>(qraw, q_nw, qn, q_bf);
  k_k_norm_rope<<<T_, 128, 0, stream>>>(kraw, k_nw, swk_bf);
  PACK(swv, swv_bf, (long)T_ * HD_);

  // compression + indexer
  k_compress<<<(NB_ * CI_ + 255) / 256, 256, 0, stream>>>(idxc, idxz, idx_b, kcomp, CI_);
  k_compress<<<(NB_ * HD_ + 255) / 256, 256, 0, stream>>>(kvcc, kvcz, kvc_b, ckv, HD_);
  k_rmsnorm_rows<<<NB_, 128, 0, stream>>>(ckv, k_nw, kvn);
  k_iscores<<<(T_ * NB_ + 255) / 256, 256, 0, stream>>>(qi, kcomp, wih, isc);
  k_topk<<<T_, 256, 0, stream>>>(isc, topidx, topval);

  // attention (sparse writes, sliding-window accumulates)
  k_sparse_attn<<<dim3(T_, H_), 128, 0, stream>>>(qn, kvn, topidx, topval, attn);
  k_sw_attn<<<dim3(T_ / 128, H_), 256, 0, stream>>>(q_bf, swk_bf, swv_bf, attn);

  // output projections
  k_attn_to_obf<<<(int)(nH / 256), 256, 0, stream>>>(attn, o_bf);
  for (int g = 0; g < G_; ++g)
    GEMM(o_bf + g * DG_, gw_bf + (long)g * DG_ * DG_, og + g * DG_, T_, DG_, DG_, HID_, DG_, HID_);
  PACK(og, og_bf, nH);
  GEMM(og_bf, fw_bf, out, T_, HID_, HID_, HID_, HID_, HID_);

#undef PACK
#undef GEMM
}
